// ModelNet10ShapeRefinement_25074019074116
// MI455X (gfx1250) — matrix operand order repaired and checked by reading, not compiled, not run on hardware
//
#include <hip/hip_runtime.h>
#include <hip/hip_bf16.h>

typedef __attribute__((ext_vector_type(16))) _Float16 v16h;
typedef __attribute__((ext_vector_type(8)))  _Float16 v8h;
typedef __attribute__((ext_vector_type(4)))  _Float16 v4h;
typedef __attribute__((ext_vector_type(8)))  float    v8f;
typedef __attribute__((ext_vector_type(4)))  float    v4f;

#define HDIM 128
#define HH   64
#define NPTS 8192
#define BATCH 32
#define TILE 128
#define FS 132    // f32 staging stride (floats): 132*4=528B -> 16B-aligned rows
#define AS 136    // f16 activation stride (halfs): 136*2=272B -> 16B-aligned frags

// LDS layout (dynamic):
//   [f32buf TILE*FS f32][act TILE*AS f16][wbuf 32KB f16 weights][smallw 640 f32]
#define SMEM_F32BUF_BYTES (TILE * FS * 4)           // 67584
#define SMEM_ACT_BYTES    (TILE * AS * 2)           // 34816
#define SMEM_WBUF_BYTES   (32768)                   // up to 128x128 f16
#define SMEM_SMALLW_BYTES (640 * 4)                 // 2560
#define SMEM_TOTAL (SMEM_F32BUF_BYTES + SMEM_ACT_BYTES + SMEM_WBUF_BYTES + SMEM_SMALLW_BYTES)

// ---------------- device helpers ----------------

__device__ __forceinline__ void stage(const float* __restrict__ g, float* l, int n, int t) {
    for (int i = t; i < n; i += 256) l[i] = g[i];
}

// Async-copy `bytes` (multiple of 4KB here) of f16 weights global -> LDS.
// Each issued instruction copies 16B per active lane (wave-level), tracked by ASYNCcnt.
__device__ __forceinline__ void stage_async_b128(const _Float16* __restrict__ g,
                                                 void* ldsdst, int bytes, int t) {
    unsigned lbase = (unsigned)(uintptr_t)ldsdst;   // low 32 bits = LDS byte address
    for (int i = t * 16; i < bytes; i += 256 * 16) {
        unsigned laddr = lbase + (unsigned)i;
        unsigned goff  = (unsigned)i;
        asm volatile("global_load_async_to_lds_b128 %0, %1, %2 offset:0"
                     :: "v"(laddr), "v"(goff), "s"(g)
                     : "memory");
    }
}

__device__ __forceinline__ void wait_async() {
    asm volatile("s_wait_asynccnt 0x0" ::: "memory");
}

// WMMA GEMM layer: out[128pts x NT*16] (f32, LDS) = act[128pts x KC*32] (f16, LDS) @ WT^T
// WT is [N][K] f16 (transposed weights), staged in LDS (wbuf).
// A-frag (16x32 f16): lanes 0-15: M=0-15, VGPR0-3 K=0..7, VGPR4-7 K=16..23;
//                     lanes 16-31: K=8..15 / 24..31.   (ISA 7.12.2)
// B-frag (32x16 f16): lane%16 = column N; same K split as A: lanes 0-15 K=0..7/16..23, lanes 16-31 K=8..15/24..31.
template<int NT, int KC>
__device__ __forceinline__ void wmma_layer(const _Float16* __restrict__ act,
                                           const _Float16* __restrict__ wt,
                                           float* __restrict__ f32buf,
                                           int lane, int m0) {
    constexpr int K = KC * 32;
    const int col = lane & 15;
    const int sel = lane >> 4;
    const _Float16* arow = act + (m0 + col) * AS + sel * 8;
#pragma unroll
    for (int nt = 0; nt < NT; ++nt) {
        v8f acc = {};
        const _Float16* bcol = wt + (nt * 16 + col) * K + sel * 8;
#pragma unroll
        for (int kc = 0; kc < KC; ++kc) {
            const _Float16* ap = arow + kc * 32;
            v8h alo = *(const v8h*)(ap);
            v8h ahi = *(const v8h*)(ap + 16);
            v16h a = __builtin_shufflevector(alo, ahi, 0,1,2,3,4,5,6,7,8,9,10,11,12,13,14,15);
            const _Float16* bp = bcol + kc * 32;
            v8h blo = *(const v8h*)(bp);
            v8h bhi = *(const v8h*)(bp + 16);
            v16h bm = __builtin_shufflevector(blo, bhi, 0,1,2,3,4,5,6,7,8,9,10,11,12,13,14,15);
            acc = __builtin_amdgcn_wmma_f32_16x16x32_f16(
                false, a, false, bm, (short)0, acc, false, false);
        }
        // C/D layout: VGPR r: lanes 0-15 -> (M=r, N=lane); lanes 16-31 -> (M=r+8, N=lane-16)
        float* orow = f32buf + (m0 + sel * 8) * FS + nt * 16 + col;
#pragma unroll
        for (int r = 0; r < 8; ++r) orow[r * FS] = acc[r];
    }
}

// LayerNorm(+bias) + LeakyReLU over W features; f32 staging -> packed f16 activations.
// Vectorized: v4f LDS loads, v4h LDS stores, fully unrolled.
template<int W>
__device__ __forceinline__ void ln_act(const float* __restrict__ f32buf,
                                       const float* __restrict__ bias,
                                       const float* __restrict__ g,
                                       const float* __restrict__ be,
                                       _Float16* __restrict__ act, int t) {
    if (t < TILE) {
        const v4f* row4 = (const v4f*)(f32buf + t * FS);
        const v4f* b4   = (const v4f*)bias;
        v4f vs = {}, vs2 = {};
#pragma unroll
        for (int i = 0; i < W / 4; ++i) {
            v4f v = row4[i] + b4[i];
            vs += v;
            vs2 += v * v;
        }
        float s  = vs[0] + vs[1] + vs[2] + vs[3];
        float s2 = vs2[0] + vs2[1] + vs2[2] + vs2[3];
        constexpr float inv = 1.0f / (float)W;
        float mean = s * inv;
        float var  = s2 * inv - mean * mean;
        float rs   = rsqrtf(var + 1e-5f);
        const v4f* g4  = (const v4f*)g;
        const v4f* be4 = (const v4f*)be;
        v4h* arow4 = (v4h*)(act + t * AS);
#pragma unroll
        for (int i = 0; i < W / 4; ++i) {
            v4f v = (row4[i] + b4[i] - mean) * rs * g4[i] + be4[i];
#pragma unroll
            for (int c = 0; c < 4; ++c) v[c] = (v[c] >= 0.f) ? v[c] : 0.2f * v[c];
            arow4[i] = __builtin_convertvector(v, v4h);
        }
    }
}

// K=3 input layer via VALU (padding 3->32 for WMMA would be 10x wasted flops)
template<int Nw>
__device__ __forceinline__ void k3_layer(float x, float y, float z,
                                         const float* __restrict__ wsm,
                                         float* __restrict__ f32buf, int t) {
    if (t < TILE) {
        v4f* row4 = (v4f*)(f32buf + t * FS);
        const v4f* w0 = (const v4f*)(wsm);
        const v4f* w1 = (const v4f*)(wsm + Nw);
        const v4f* w2 = (const v4f*)(wsm + 2 * Nw);
#pragma unroll
        for (int i = 0; i < Nw / 4; ++i)
            row4[i] = x * w0[i] + y * w1[i] + z * w2[i];
    }
}

// 3-wide output head: o = act[K] @ W[K][3] + b[3]
template<int Kw>
__device__ __forceinline__ void out3(const _Float16* __restrict__ act,
                                     const float* __restrict__ wsm,
                                     const float* __restrict__ bsm,
                                     float* o, int t) {
    if (t < TILE) {
        const v4h* arow4 = (const v4h*)(act + t * AS);
        float o0 = bsm[0], o1 = bsm[1], o2 = bsm[2];
#pragma unroll
        for (int i = 0; i < Kw / 4; ++i) {
            v4h a4 = arow4[i];
#pragma unroll
            for (int c = 0; c < 4; ++c) {
                float a = (float)a4[c];
                int k = i * 4 + c;
                o0 += a * wsm[k * 3 + 0];
                o1 += a * wsm[k * 3 + 1];
                o2 += a * wsm[k * 3 + 2];
            }
        }
        o[0] = o0; o[1] = o1; o[2] = o2;
    } else {
        o[0] = o[1] = o[2] = 0.f;
    }
}

// ---------------- prep kernel: f32 weights -> transposed f16; fold ce into bias ----

__global__ void shape_refine_prep(const float* __restrict__ pr_w2,
                                  const float* __restrict__ cr_w1,
                                  const float* __restrict__ cr_w2,
                                  const float* __restrict__ sp_w2,
                                  const float* __restrict__ cat_emb,
                                  const float* __restrict__ cr_b1,
                                  const int* __restrict__ cids,
                                  _Float16* __restrict__ wt_pr2,
                                  _Float16* __restrict__ wt_cr1a,
                                  _Float16* __restrict__ wt_cr2,
                                  _Float16* __restrict__ wt_sp2,
                                  float* __restrict__ ce_bias) {
    int tid = blockIdx.x * blockDim.x + threadIdx.x;
    int stride = gridDim.x * blockDim.x;
    for (int i = tid; i < HDIM * HDIM; i += stride) {
        int n = i >> 7, k = i & (HDIM - 1);
        wt_pr2[n * HDIM + k]  = (_Float16)pr_w2[k * HDIM + n];
        wt_cr1a[n * HDIM + k] = (_Float16)cr_w1[k * HDIM + n];
    }
    for (int i = tid; i < HH * HDIM; i += stride) {
        int n = i >> 7, k = i & (HDIM - 1);
        wt_cr2[n * HDIM + k] = (_Float16)cr_w2[k * HH + n];
    }
    for (int i = tid; i < 10 * HH * HDIM; i += stride) {
        int e = i / (HH * HDIM);
        int r = i % (HH * HDIM);
        int n = r >> 7, k = r & (HDIM - 1);
        wt_sp2[e * HH * HDIM + n * HDIM + k] = (_Float16)sp_w2[e * HDIM * HH + k * HH + n];
    }
    // ce_bias[b][n] = cr_b1[n] + sum_j cat_emb[cid[b]][j] * cr_w1[128+j][n]
    for (int i = tid; i < BATCH * HDIM; i += stride) {
        int bb = i >> 7, n = i & (HDIM - 1);
        int c = cids[bb];
        float acc = cr_b1[n];
        for (int j = 0; j < HDIM; ++j)
            acc += cat_emb[c * HDIM + j] * cr_w1[(HDIM + j) * HDIM + n];
        ce_bias[i] = acc;
    }
}

// ---------------- main kernel ----------------

struct Params {
    const float* points; const int* cids;
    const float *pr_w1, *pr_b1, *pr_g1, *pr_be1, *pr_b2, *pr_g2, *pr_be2;
    const float *cr_g1, *cr_be1, *cr_b2, *cr_g2, *cr_be2, *cr_w3, *cr_b3;
    const float *sy_w1, *sy_b1, *sy_g1, *sy_be1, *sy_w2, *sy_b2;
    const float *sp_w1, *sp_b1, *sp_g1, *sp_be1, *sp_b2, *sp_g2, *sp_be2, *sp_w3, *sp_b3;
    const _Float16 *wt_pr2, *wt_cr1a, *wt_cr2, *wt_sp2;
    const float* ce_bias;
    float* out;
};

__global__ __launch_bounds__(256) void shape_refine_main(Params p) {
    extern __shared__ __align__(16) char smem[];
    float*    f32buf = (float*)smem;
    _Float16* act    = (_Float16*)(smem + SMEM_F32BUF_BYTES);
    _Float16* wbuf   = (_Float16*)(smem + SMEM_F32BUF_BYTES + SMEM_ACT_BYTES);
    float*    wsm    = (float*)(smem + SMEM_F32BUF_BYTES + SMEM_ACT_BYTES + SMEM_WBUF_BYTES);
    float*    bsm    = wsm + 600;

    const int t    = threadIdx.x;
    const int lane = t & 31;
    const int m0   = (t >> 5) * 16;      // 8 waves * 16 points = 128 points
    const int b    = blockIdx.x >> 6;    // 64 tiles per batch
    const int p0   = (blockIdx.x & 63) * TILE;
    const int cid  = p.cids[b];

    // kick off async staging of pr2 weights immediately (overlaps pr1)
    stage_async_b128(p.wt_pr2, wbuf, HDIM * HDIM * 2, t);

    float x = 0.f, y = 0.f, z = 0.f;
    if (t < TILE) {
        const float* pt = p.points + ((size_t)b * NPTS + p0 + t) * 3;
        x = pt[0]; y = pt[1]; z = pt[2];
    }

    // ---- pr1: points(3) -> 128, VALU ----
    stage(p.pr_w1, wsm, 3 * HDIM, t);
    __syncthreads();
    k3_layer<HDIM>(x, y, z, wsm, f32buf, t);
    __syncthreads();
    ln_act<HDIM>(f32buf, p.pr_b1, p.pr_g1, p.pr_be1, act, t);
    wait_async();
    __syncthreads();

    // ---- pr2: 128 -> 128, WMMA (weights in LDS) ----
    wmma_layer<8, 4>(act, wbuf, f32buf, lane, m0);
    __syncthreads();
    stage_async_b128(p.wt_cr1a, wbuf, HDIM * HDIM * 2, t);   // prefetch cr1
    ln_act<HDIM>(f32buf, p.pr_b2, p.pr_g2, p.pr_be2, act, t);
    wait_async();
    __syncthreads();

    // ---- cr1: pf(128) -> 128, category embedding folded into per-batch bias ----
    wmma_layer<8, 4>(act, wbuf, f32buf, lane, m0);
    __syncthreads();
    stage_async_b128(p.wt_cr2, wbuf, HH * HDIM * 2, t);      // prefetch cr2
    ln_act<HDIM>(f32buf, p.ce_bias + b * HDIM, p.cr_g1, p.cr_be1, act, t);
    wait_async();
    __syncthreads();

    // ---- cr2: 128 -> 64, WMMA ----
    wmma_layer<4, 4>(act, wbuf, f32buf, lane, m0);
    __syncthreads();
    stage_async_b128(p.wt_sp2 + cid * (HH * HDIM), wbuf, HH * HDIM * 2, t); // prefetch sp2
    ln_act<HH>(f32buf, p.cr_b2, p.cr_g2, p.cr_be2, act, t);
    __syncthreads();

    // ---- offsets = h @ cr_w3 + cr_b3 ----
    stage(p.cr_w3, wsm, HH * 3, t);
    stage(p.cr_b3, bsm, 3, t);
    __syncthreads();
    float off[3];
    out3<HH>(act, wsm, bsm, off, t);
    __syncthreads();

    // ---- sp branch (per-category prior) ----
    stage(p.sp_w1 + cid * 3 * HDIM, wsm, 3 * HDIM, t);
    __syncthreads();
    k3_layer<HDIM>(x, y, z, wsm, f32buf, t);
    __syncthreads();
    ln_act<HDIM>(f32buf, p.sp_b1 + cid * HDIM, p.sp_g1 + cid * HDIM,
                 p.sp_be1 + cid * HDIM, act, t);
    wait_async();
    __syncthreads();

    wmma_layer<4, 4>(act, wbuf, f32buf, lane, m0);
    __syncthreads();
    ln_act<HH>(f32buf, p.sp_b2 + cid * HH, p.sp_g2 + cid * HH,
               p.sp_be2 + cid * HH, act, t);
    __syncthreads();

    stage(p.sp_w3 + cid * HH * 3, wsm, HH * 3, t);
    stage(p.sp_b3 + cid * 3, bsm, 3, t);
    __syncthreads();
    float pri[3];
    out3<HH>(act, wsm, bsm, pri, t);
    off[0] += 0.5f * pri[0];
    off[1] += 0.5f * pri[1];
    off[2] += 0.5f * pri[2];
    __syncthreads();

    // ---- sy (symmetry) branch ----
    stage(p.sy_w1, wsm, 3 * HH, t);
    __syncthreads();
    k3_layer<HH>(-x, y, z, wsm, f32buf, t);   // mx = xyz * [-1,1,1]
    __syncthreads();
    ln_act<HH>(f32buf, p.sy_b1, p.sy_g1, p.sy_be1, act, t);
    __syncthreads();
    stage(p.sy_w2, wsm, HH * 3, t);
    stage(p.sy_b2, bsm, 3, t);
    __syncthreads();
    float s[3];
    out3<HH>(act, wsm, bsm, s, t);
    s[0] = -s[0];                              // * flip

    if (cid == 2 || cid == 8) {
        off[0] = (off[0] + s[0]) * 0.5f;
        off[1] = (off[1] + s[1]) * 0.5f;
        off[2] = (off[2] + s[2]) * 0.5f;
    }

    if (t < TILE) {
        float* op = p.out + ((size_t)b * NPTS + p0 + t) * 3;
        op[0] = off[0]; op[1] = off[1]; op[2] = off[2];
    }
}

// ---------------- launcher ----------------

extern "C" void kernel_launch(void* const* d_in, const int* in_sizes, int n_in,
                              void* d_out, int out_size, void* d_ws, size_t ws_size,
                              hipStream_t stream) {
    const float* points  = (const float*)d_in[0];
    const int*   cids    = (const int*)  d_in[1];
    const float* pr_w1   = (const float*)d_in[2];
    const float* pr_b1   = (const float*)d_in[3];
    const float* pr_g1   = (const float*)d_in[4];
    const float* pr_be1  = (const float*)d_in[5];
    const float* pr_w2   = (const float*)d_in[6];
    const float* pr_b2   = (const float*)d_in[7];
    const float* pr_g2   = (const float*)d_in[8];
    const float* pr_be2  = (const float*)d_in[9];
    const float* cat_emb = (const float*)d_in[10];
    const float* cr_w1   = (const float*)d_in[11];
    const float* cr_b1   = (const float*)d_in[12];
    const float* cr_g1   = (const float*)d_in[13];
    const float* cr_be1  = (const float*)d_in[14];
    const float* cr_w2   = (const float*)d_in[15];
    const float* cr_b2   = (const float*)d_in[16];
    const float* cr_g2   = (const float*)d_in[17];
    const float* cr_be2  = (const float*)d_in[18];
    const float* cr_w3   = (const float*)d_in[19];
    const float* cr_b3   = (const float*)d_in[20];
    const float* sy_w1   = (const float*)d_in[21];
    const float* sy_b1   = (const float*)d_in[22];
    const float* sy_g1   = (const float*)d_in[23];
    const float* sy_be1  = (const float*)d_in[24];
    const float* sy_w2   = (const float*)d_in[25];
    const float* sy_b2   = (const float*)d_in[26];
    const float* sp_w1   = (const float*)d_in[27];
    const float* sp_b1   = (const float*)d_in[28];
    const float* sp_g1   = (const float*)d_in[29];
    const float* sp_be1  = (const float*)d_in[30];
    const float* sp_w2   = (const float*)d_in[31];
    const float* sp_b2   = (const float*)d_in[32];
    const float* sp_g2   = (const float*)d_in[33];
    const float* sp_be2  = (const float*)d_in[34];
    const float* sp_w3   = (const float*)d_in[35];
    const float* sp_b3   = (const float*)d_in[36];

    // scratch layout (all 4KB-aligned offsets)
    char* ws = (char*)d_ws;
    _Float16* wt_pr2  = (_Float16*)(ws + 0);        // 128*128 f16 = 32768 B
    _Float16* wt_cr1a = (_Float16*)(ws + 32768);    // 128*128 f16 = 32768 B
    _Float16* wt_cr2  = (_Float16*)(ws + 65536);    // 64*128  f16 = 16384 B
    _Float16* wt_sp2  = (_Float16*)(ws + 81920);    // 10*64*128 f16 = 163840 B
    float*    ce_bias = (float*)  (ws + 245760);    // 32*128 f32 = 16384 B

    shape_refine_prep<<<128, 256, 0, stream>>>(
        pr_w2, cr_w1, cr_w2, sp_w2, cat_emb, cr_b1, cids,
        wt_pr2, wt_cr1a, wt_cr2, wt_sp2, ce_bias);

    Params P;
    P.points = points; P.cids = cids;
    P.pr_w1 = pr_w1; P.pr_b1 = pr_b1; P.pr_g1 = pr_g1; P.pr_be1 = pr_be1;
    P.pr_b2 = pr_b2; P.pr_g2 = pr_g2; P.pr_be2 = pr_be2;
    P.cr_g1 = cr_g1; P.cr_be1 = cr_be1; P.cr_b2 = cr_b2; P.cr_g2 = cr_g2; P.cr_be2 = cr_be2;
    P.cr_w3 = cr_w3; P.cr_b3 = cr_b3;
    P.sy_w1 = sy_w1; P.sy_b1 = sy_b1; P.sy_g1 = sy_g1; P.sy_be1 = sy_be1;
    P.sy_w2 = sy_w2; P.sy_b2 = sy_b2;
    P.sp_w1 = sp_w1; P.sp_b1 = sp_b1; P.sp_g1 = sp_g1; P.sp_be1 = sp_be1;
    P.sp_b2 = sp_b2; P.sp_g2 = sp_g2; P.sp_be2 = sp_be2;
    P.sp_w3 = sp_w3; P.sp_b3 = sp_b3;
    P.wt_pr2 = wt_pr2; P.wt_cr1a = wt_cr1a; P.wt_cr2 = wt_cr2; P.wt_sp2 = wt_sp2;
    P.ce_bias = ce_bias;
    P.out = (float*)d_out;

    dim3 grid(BATCH * (NPTS / TILE));   // 32 * 64 = 2048 blocks
    shape_refine_main<<<grid, 256, SMEM_TOTAL, stream>>>(P);
}
